// NetVLAD_12326556139531
// MI455X (gfx1250) — matrix operand order repaired and checked by reading, not compiled, not run on hardware
//
#include <hip/hip_runtime.h>
#include <hip/hip_bf16.h>
#include <math.h>

// ---------------------------------------------------------------------------
// NetVLAD fused kernel for gfx1250 (MI455X), wave32 + v_wmma_f32_16x16x32_bf16
// + async global->LDS double-buffered staging (ASYNCcnt) when available.
// ---------------------------------------------------------------------------

typedef __attribute__((ext_vector_type(16))) __bf16 v16bf;
typedef __attribute__((ext_vector_type(8)))  float  v8f;
typedef __attribute__((ext_vector_type(4)))  int    int4v;

union FragBF {
    v16bf v;
    uint4 q[2];
};

// Detect CDNA5 async load-to-LDS builtin (device pass only; host pass falls back)
#if defined(__has_builtin)
#if __has_builtin(__builtin_amdgcn_global_load_async_to_lds_b128)
#define HAS_ASYNC_LDS 1
#endif
#endif
#ifndef HAS_ASYNC_LDS
#define HAS_ASYNC_LDS 0
#endif

// Problem constants
constexpr int N_ = 32;
constexpr int C_ = 512;
constexpr int P_ = 3136;   // 56*56
constexpr int K_ = 64;

constexpr int PT   = 64;   // pixels per tile
constexpr int TPC  = 7;    // tiles per chunk  (7 chunks * 7 tiles * 64 px = 3136)
constexpr int NCHK = 7;

// LDS layout (bytes); strides padded for bank spread + 16B fragment alignment
constexpr int XPC_STR = 520;  // x^T  [p][c], bf16; 1040B rows (65*16)
constexpr int XCP_STR = 72;   // x    [c][p], bf16; 144B rows (9*16)
constexpr int A_STR   = 72;   // a    [k][p], bf16
constexpr int LG_STR  = 65;   // logits [p][k], f32
constexpr int STG_STR = 68;   // staging [c][p] f32; 272B rows (17*16)

constexpr int OFF_XPC  = 0;                               // 64*520*2  = 66560
constexpr int OFF_XCP  = OFF_XPC + PT * XPC_STR * 2;      // 512*72*2  = 73728
constexpr int OFF_LG   = OFF_XCP + C_ * XCP_STR * 2;      // 64*65*4   = 16640
constexpr int OFF_A    = OFF_LG  + PT * LG_STR * 4;       // 64*72*2   = 9216
constexpr int OFF_ASUM = OFF_A   + K_ * A_STR * 2;        // 64*4      = 256
constexpr int OFF_STG  = OFF_ASUM + K_ * 4;               // 512*68*4  = 139264
constexpr int SMEM_BYTES = OFF_STG + C_ * STG_STR * 4;    // = 305664 (< 320KB/WGP)

// Workspace layout (bytes)
constexpr size_t WS_ACC  = 0;                                  // N*C*K f32 = 4 MiB
constexpr size_t WS_ASUM = WS_ACC  + (size_t)N_ * C_ * K_ * 4; // N*K f32
constexpr size_t WS_GSUM = WS_ASUM + (size_t)N_ * K_ * 4;      // N f32
constexpr size_t WS_WBF  = WS_GSUM + (size_t)N_ * 4;           // K*C bf16

__global__ void zero_f32_kernel(float* __restrict__ p, int n) {
    int i = blockIdx.x * blockDim.x + threadIdx.x;
    if (i < n) p[i] = 0.0f;
}

__global__ void cvt_bf16_kernel(const float* __restrict__ w, __bf16* __restrict__ o, int n) {
    int i = blockIdx.x * blockDim.x + threadIdx.x;
    if (i < n) o[i] = (__bf16)w[i];
}

#if HAS_ASYNC_LDS
typedef __attribute__((address_space(1))) int4v* g4ptr;
typedef __attribute__((address_space(3))) int4v* l4ptr;

__device__ __forceinline__ void async_cp16(const float* g, float* l) {
    // builtin signature (from hipcc diagnostic): arg0 = int4* addrspace(1)
    __builtin_amdgcn_global_load_async_to_lds_b128(
        (g4ptr)(const __attribute__((address_space(1))) float*)g,
        (l4ptr)(__attribute__((address_space(3))) float*)l,
        0, 0);
}

__device__ __forceinline__ void wait_async0() {
#if __has_builtin(__builtin_amdgcn_s_wait_asynccnt)
    __builtin_amdgcn_s_wait_asynccnt(0);
#else
    asm volatile("s_wait_asynccnt 0x0" ::: "memory");
#endif
}
#endif

__global__ __launch_bounds__(512)
void netvlad_main_kernel(const float* __restrict__ x,
                         const __bf16* __restrict__ wbf,   // conv_w as bf16, [K][C] row-major
                         const float* __restrict__ conv_b, // [K]
                         float* __restrict__ acc,          // [N][C][K] f32 (atomic accum)
                         float* __restrict__ asum_g)       // [N][K]    f32 (atomic accum)
{
    extern __shared__ char smem[];
    __bf16* x_pc   = (__bf16*)(smem + OFF_XPC);   // [p][c]
    __bf16* x_cp   = (__bf16*)(smem + OFF_XCP);   // [c][p]
    float*  lg     = (float*) (smem + OFF_LG);    // [p][k]
    __bf16* a_sm   = (__bf16*)(smem + OFF_A);     // [k][p]
    float*  asum_l = (float*) (smem + OFF_ASUM);  // [k]
#if HAS_ASYNC_LDS
    float*  stg    = (float*) (smem + OFF_STG);   // [c][p] f32 prefetch buffer
#endif

    const int n     = blockIdx.y;
    const int chunk = blockIdx.x;
    const int tid   = threadIdx.x;
    const int wv    = tid >> 5;
    const int lane  = tid & 31;
    const int l16   = lane & 15;
    const int hh    = lane >> 4;      // 0: lanes 0-15, 1: lanes 16-31

    // step-A role: wave wv computes logits^T tile (pt_a, kt_a)
    const int pt_a = wv >> 2;
    const int kt_a = wv & 3;
    const float bias = conv_b[kt_a * 16 + l16];

    // step-C accumulators: c-tiles {wv, wv+16}, k-tiles 0..3
    const v8f vzero = {0.f, 0.f, 0.f, 0.f, 0.f, 0.f, 0.f, 0.f};
    v8f accf[2][4];
#pragma unroll
    for (int cs = 0; cs < 2; ++cs)
#pragma unroll
        for (int kt = 0; kt < 4; ++kt) accf[cs][kt] = vzero;

    if (tid < K_) asum_l[tid] = 0.0f;

    const float* xn = x + (size_t)n * C_ * P_;
    const int pbase0 = chunk * TPC * PT;

#if HAS_ASYNC_LDS
    // prologue: async-prefetch tile 0 into LDS staging buffer
#pragma unroll
    for (int it = 0; it < 16; ++it) {
        const int f  = it * 512 + tid;
        const int c  = f >> 4;
        const int p4 = f & 15;
        async_cp16(xn + (size_t)c * P_ + pbase0 + p4 * 4,
                   stg + c * STG_STR + p4 * 4);
    }
#endif

    for (int tile = 0; tile < TPC; ++tile) {
        const int pbase = pbase0 + tile * PT;

        // ---- convert staged tile (f32) -> bf16 in both LDS layouts ----
#if HAS_ASYNC_LDS
        wait_async0();   // this wave's staged bytes are in LDS
#endif
#pragma unroll
        for (int it = 0; it < 16; ++it) {
            const int f  = it * 512 + tid;     // float4 index over 512x64 tile
            const int c  = f >> 4;
            const int p4 = f & 15;
#if HAS_ASYNC_LDS
            const float4 v = *(const float4*)(stg + c * STG_STR + p4 * 4);
#else
            const float4 v = *(const float4*)(xn + (size_t)c * P_ + pbase + p4 * 4);
#endif
            union { __bf16 h[4]; uint2 u; } pk;
            pk.h[0] = (__bf16)v.x; pk.h[1] = (__bf16)v.y;
            pk.h[2] = (__bf16)v.z; pk.h[3] = (__bf16)v.w;
            // [c][p] packed 8B store
            *(uint2*)(x_cp + c * XCP_STR + p4 * 4) = pk.u;
            // [p][c] scattered bf16 stores (staging only)
            x_pc[(p4 * 4 + 0) * XPC_STR + c] = pk.h[0];
            x_pc[(p4 * 4 + 1) * XPC_STR + c] = pk.h[1];
            x_pc[(p4 * 4 + 2) * XPC_STR + c] = pk.h[2];
            x_pc[(p4 * 4 + 3) * XPC_STR + c] = pk.h[3];
        }
        __syncthreads();   // x_pc/x_cp visible to all waves; own ds reads drained

#if HAS_ASYNC_LDS
        // kick off async prefetch of next tile (overlaps with compute below)
        if (tile + 1 < TPC) {
            const int pnext = pbase + PT;
#pragma unroll
            for (int it = 0; it < 16; ++it) {
                const int f  = it * 512 + tid;
                const int c  = f >> 4;
                const int p4 = f & 15;
                async_cp16(xn + (size_t)c * P_ + pnext + p4 * 4,
                           stg + c * STG_STR + p4 * 4);
            }
        }
#endif

        // ---- step A: logits^T(p,k) = sum_c x^T(p,c) * conv_w^T(c,k) ----
        {
            v8f lacc = vzero;
#pragma unroll
            for (int ct = 0; ct < 16; ++ct) {
                const int cb = ct * 32;
                FragBF fa, fb;
                // A = x^T, row-major [p][c]: e0-7 @ cb+hh*8, e8-15 @ cb+16+hh*8
                const __bf16* ap = x_pc + (pt_a * 16 + l16) * XPC_STR + cb + hh * 8;
                fa.q[0] = *(const uint4*)ap;
                fa.q[1] = *(const uint4*)(ap + 16);
                // B = conv_w^T column-major == conv_w row-major [k][c]:
                // e0-7 @ cb+hh*8, e8-15 @ cb+16+hh*8
                const __bf16* bp = wbf + (size_t)(kt_a * 16 + l16) * C_ + cb + hh * 8;
                fb.q[0] = *(const uint4*)bp;
                fb.q[1] = *(const uint4*)(bp + 16);
                lacc = __builtin_amdgcn_wmma_f32_16x16x32_bf16(
                    false, fa.v, false, fb.v, (short)0, lacc, false, false);
            }
            // D layout: vgpr j -> M = j + 8*hh (=p-offset), N = l16 (=k-offset)
#pragma unroll
            for (int j = 0; j < 8; ++j) {
                const int p = pt_a * 16 + j + hh * 8;
                lg[p * LG_STR + kt_a * 16 + l16] = lacc[j] + bias;
            }
        }
        __syncthreads();

        // ---- softmax over K=64 per pixel: thread -> (p=tid>>3, seg=tid&7) ----
        {
            const int p = tid >> 3, seg = tid & 7;
            const float* row = lg + p * LG_STR + seg * 8;
            float v[8];
            float m = -INFINITY;
#pragma unroll
            for (int i = 0; i < 8; ++i) { v[i] = row[i]; m = fmaxf(m, v[i]); }
            m = fmaxf(m, __shfl_xor(m, 1));
            m = fmaxf(m, __shfl_xor(m, 2));
            m = fmaxf(m, __shfl_xor(m, 4));
            float s = 0.0f;
#pragma unroll
            for (int i = 0; i < 8; ++i) { v[i] = __expf(v[i] - m); s += v[i]; }
            s += __shfl_xor(s, 1);
            s += __shfl_xor(s, 2);
            s += __shfl_xor(s, 4);
            const float inv = 1.0f / s;
#pragma unroll
            for (int i = 0; i < 8; ++i)
                a_sm[(seg * 8 + i) * A_STR + p] = (__bf16)(v[i] * inv);
        }
        __syncthreads();

        // ---- a column-sum partials: thread -> (k=tid>>3, seg=tid&7) ----
        {
            const int k = tid >> 3, seg = tid & 7;
            const __bf16* row = a_sm + k * A_STR + seg * 8;
            float s = 0.0f;
#pragma unroll
            for (int i = 0; i < 8; ++i) s += (float)row[i];
            s += __shfl_xor(s, 1);
            s += __shfl_xor(s, 2);
            s += __shfl_xor(s, 4);
            if (seg == 0) asum_l[k] += s;   // unique writer per k
        }

        // ---- step C: vlad(c,k) += sum_p x(c,p) * a^T(p,k) ----
#pragma unroll
        for (int half = 0; half < 2; ++half) {
            const int pb = half * 32;
            FragBF fB[4];
#pragma unroll
            for (int kt = 0; kt < 4; ++kt) {
                // B = a^T column-major == a row-major [k][p]: e0-7 @ pb+hh*8, e8-15 @ pb+16+hh*8
                const __bf16* bp = a_sm + (kt * 16 + l16) * A_STR + pb + hh * 8;
                fB[kt].q[0] = *(const uint4*)bp;
                fB[kt].q[1] = *(const uint4*)(bp + 16);
            }
#pragma unroll
            for (int cs = 0; cs < 2; ++cs) {
                const int ct = wv + 16 * cs;
                FragBF fA;
                // A = x row-major [c][p]: e0-7 @ pb+hh*8, e8-15 @ pb+16+hh*8
                const __bf16* ap = x_cp + (ct * 16 + l16) * XCP_STR + pb + hh * 8;
                fA.q[0] = *(const uint4*)ap;
                fA.q[1] = *(const uint4*)(ap + 16);
#pragma unroll
                for (int kt = 0; kt < 4; ++kt)
                    accf[cs][kt] = __builtin_amdgcn_wmma_f32_16x16x32_bf16(
                        false, fA.v, false, fB[kt].v, (short)0, accf[cs][kt], false, false);
            }
        }
        __syncthreads();   // protect x_pc/x_cp/lg/a_sm before next tile's convert
    } // tiles

    // ---- flush accumulators to global (atomic f32 add) ----
    float* accn = acc + (size_t)n * C_ * K_;
#pragma unroll
    for (int cs = 0; cs < 2; ++cs) {
#pragma unroll
        for (int kt = 0; kt < 4; ++kt) {
#pragma unroll
            for (int j = 0; j < 8; ++j) {
                const int c = (wv + 16 * cs) * 16 + j + hh * 8;
                const int k = kt * 16 + l16;
                atomicAdd(accn + c * K_ + k, accf[cs][kt][j]);
            }
        }
    }
    if (tid < K_) atomicAdd(asum_g + n * K_ + tid, asum_l[tid]);
}

// vlad = acc - cent_r * asum ; intra-normalize over K per (n,c) row
__global__ __launch_bounds__(256)
void finalize1_kernel(const float* __restrict__ acc,
                      const float* __restrict__ asum,
                      const float* __restrict__ cent,   // centroids flat (K*C); cent_r[c][k] = flat[c*K+k]
                      float* __restrict__ out,
                      float* __restrict__ gsum)
{
    const int r    = blockIdx.x * 8 + (threadIdx.x >> 5);  // row over N*C
    const int lane = threadIdx.x & 31;
    const int n = r >> 9;           // / C_
    const int c = r & (C_ - 1);
    const float* arow = acc + (size_t)r * K_;
    const float* as   = asum + n * K_;
    const float* ce   = cent + c * K_;
    float v0 = arow[lane]      - ce[lane]      * as[lane];
    float v1 = arow[lane + 32] - ce[lane + 32] * as[lane + 32];
    float ss = v0 * v0 + v1 * v1;
    ss += __shfl_xor(ss, 16);
    ss += __shfl_xor(ss, 8);
    ss += __shfl_xor(ss, 4);
    ss += __shfl_xor(ss, 2);
    ss += __shfl_xor(ss, 1);
    const float nrm   = sqrtf(ss);
    const float denom = fmaxf(nrm, 1e-12f);
    v0 /= denom;
    v1 /= denom;
    out[(size_t)r * K_ + lane]      = v0;
    out[(size_t)r * K_ + lane + 32] = v1;
    if (lane == 0) atomicAdd(gsum + n, ss / (denom * denom));
}

// global L2 normalization per batch
__global__ void finalize2_kernel(float* __restrict__ out, const float* __restrict__ gsum) {
    const int i = blockIdx.x * blockDim.x + threadIdx.x;  // over N*C*K
    const int n = i >> 15;  // / (C_*K_)
    out[i] /= fmaxf(sqrtf(gsum[n]), 1e-12f);
}

extern "C" void kernel_launch(void* const* d_in, const int* in_sizes, int n_in,
                              void* d_out, int out_size, void* d_ws, size_t ws_size,
                              hipStream_t stream) {
    const float* x      = (const float*)d_in[0];  // (N,C,H,W)
    const float* cent   = (const float*)d_in[1];  // (K,C)
    const float* conv_w = (const float*)d_in[2];  // (K,C)
    const float* conv_b = (const float*)d_in[3];  // (K,)
    float* out = (float*)d_out;

    char* ws = (char*)d_ws;
    float*  acc  = (float*)(ws + WS_ACC);
    float*  asum = (float*)(ws + WS_ASUM);
    float*  gsum = (float*)(ws + WS_GSUM);
    __bf16* wbf  = (__bf16*)(ws + WS_WBF);

    // zero acc + asum + gsum (contiguous in ws)
    const int nz = N_ * C_ * K_ + N_ * K_ + N_;
    zero_f32_kernel<<<(nz + 255) / 256, 256, 0, stream>>>((float*)ws, nz);

    // convert conv_w to bf16 once per launch
    cvt_bf16_kernel<<<(K_ * C_ + 255) / 256, 256, 0, stream>>>(conv_w, wbf, K_ * C_);

    dim3 grid(NCHK, N_);
    // NOTE: always reserve the max LDS (host pass can't see device __has_builtin)
    netvlad_main_kernel<<<grid, 512, SMEM_BYTES, stream>>>(x, wbf, conv_b, acc, asum);

    finalize1_kernel<<<(N_ * C_) / 8, 256, 0, stream>>>(acc, asum, cent, out, gsum);
    finalize2_kernel<<<(N_ * C_ * K_) / 256, 256, 0, stream>>>(out, gsum);
}
